// CausalSelfAttention_18519898980532
// MI455X (gfx1250) — matrix operand order repaired and checked by reading, not compiled, not run on hardware
//
#include <hip/hip_runtime.h>
#include <stdint.h>

typedef __attribute__((ext_vector_type(16))) __bf16 v16bf;
typedef __attribute__((ext_vector_type(8)))  float  v8f;

#define B_    2
#define S_    2048
#define HID_  768
#define KV_   4
#define G_    3
#define D_    64
#define MTOT  (B_ * S_)          // 4096 rows for all GEMMs
#define KVD_  (KV_ * D_)         // 256
#define SCALE_ 0.125f            // 64^-0.5
#define NEGINF_ (-1.0e30f)

#define WMMA_BF16(a, b, c) \
  __builtin_amdgcn_wmma_f32_16x16x32_bf16(false, (a), false, (b), (short)0, (c), false, false)

// ---------- helpers ----------------------------------------------------------

static __device__ __forceinline__ uint16_t f2bf(float f) {
  uint32_t u = __builtin_bit_cast(uint32_t, f);
  u += 0x7FFFu + ((u >> 16) & 1u);        // round-to-nearest-even
  return (uint16_t)(u >> 16);
}

union FragBF { uint32_t u[8]; v16bf v; };

// A operand: 16x32 bf16 tile from row-major storage, base = &src[row0][k0].
// ISA 7.12.2 (16-bit A 16x32): lane = g*16+m ; VGPR j<4: K = 8g+2j ; j>=4: K = 16+8g+2(j-4)
static __device__ __forceinline__ v16bf load_a_frag(const uint16_t* base, int ld) {
  const int lid = threadIdx.x & 31;
  const int g = lid >> 4, m = lid & 15;
  const uint16_t* row = base + (size_t)m * ld;
  FragBF f;
#pragma unroll
  for (int j = 0; j < 4; ++j)
    f.u[j] = *(const uint32_t*)(row + g * 8 + 2 * j);
#pragma unroll
  for (int j = 0; j < 4; ++j)
    f.u[4 + j] = *(const uint32_t*)(row + 16 + g * 8 + 2 * j);
  return f.v;
}

// B operand: 32x16 bf16 tile (K x N) read from TRANSPOSED storage [N][K],
// base = &srcT[n0][k0]. Same K order as A: lane = g*16+n ; VGPR j<4: K = 8g+2j ;
// j>=4: K = 16+8g+2(j-4).
static __device__ __forceinline__ v16bf load_b_frag(const uint16_t* baseT, int ld) {
  const int lid = threadIdx.x & 31;
  const int g = lid >> 4, n = lid & 15;
  const uint16_t* row = baseT + (size_t)n * ld;
  FragBF f;
#pragma unroll
  for (int j = 0; j < 8; ++j)
    f.u[j] = *(const uint32_t*)(row + (j >> 2) * 16 + g * 8 + 2 * (j & 3));
  return f.v;
}

// Row-wise reductions over a 16-lane group (C-tile rows live per half-wave).
static __device__ __forceinline__ float rowmax16(float x) {
#pragma unroll
  for (int m = 1; m < 16; m <<= 1) x = fmaxf(x, __shfl_xor(x, m, 16));
  return x;
}
static __device__ __forceinline__ float rowsum16(float x) {
#pragma unroll
  for (int m = 1; m < 16; m <<= 1) x += __shfl_xor(x, m, 16);
  return x;
}

// ---------- conversion kernels ----------------------------------------------

__global__ __launch_bounds__(256, 1) void cvt_bf16(const float* __restrict__ in,
                                                   uint16_t* __restrict__ out, int n) {
  int i = blockIdx.x * blockDim.x + threadIdx.x;
  if (i < n) out[i] = f2bf(in[i]);
}

// W [K][N] f32 -> WT [N][K] bf16
__global__ __launch_bounds__(256, 1) void transpose_cvt(const float* __restrict__ W,
                                                        uint16_t* __restrict__ WT,
                                                        int K, int N) {
  int i = blockIdx.x * blockDim.x + threadIdx.x;
  if (i >= K * N) return;
  int k = i / N, n = i - k * N;
  WT[(size_t)n * K + k] = f2bf(W[i]);
}

// ---------- WMMA GEMM: C[M,N] = A[M,K](bf16) * BT[N,K](bf16)^T + bias --------
// One wave computes a 32x32 C block (2x2 WMMA tiles, A/B fragments reused 2x).
// MODE 0: store bf16 [M,N]   (Q, K projections)
// MODE 1: store bf16 [N,M]   (V projection, pre-transposed for PV WMMA)
// MODE 2: store f32  [M,N]   (final output projection)
template <int MODE>
__global__ __launch_bounds__(256, 1)
void gemm_wmma(const uint16_t* __restrict__ A, const uint16_t* __restrict__ BT,
               const float* __restrict__ bias, void* __restrict__ out,
               int M, int N, int K) {
  const int wv = (int)((blockIdx.x * blockDim.x + threadIdx.x) >> 5);
  const int tn = N >> 5;                      // 32-wide n tiles
  const int n_t = wv % tn, m_t = wv / tn;

  const uint16_t* A0 = A  + (size_t)(m_t * 32) * K;
  const uint16_t* A1 = A0 + (size_t)16 * K;
  const uint16_t* B0 = BT + (size_t)(n_t * 32) * K;
  const uint16_t* B1 = B0 + (size_t)16 * K;

  v8f c00 = {}, c01 = {}, c10 = {}, c11 = {};
  for (int k0 = 0; k0 < K; k0 += 32) {
    v16bf a0 = load_a_frag(A0 + k0, K);
    v16bf a1 = load_a_frag(A1 + k0, K);
    v16bf b0 = load_b_frag(B0 + k0, K);
    v16bf b1 = load_b_frag(B1 + k0, K);
    c00 = WMMA_BF16(a0, b0, c00);
    c01 = WMMA_BF16(a0, b1, c01);
    c10 = WMMA_BF16(a1, b0, c10);
    c11 = WMMA_BF16(a1, b1, c11);
  }

  const int lid  = threadIdx.x & 31;
  const int half = lid >> 4, n = lid & 15;
  const float bv0 = bias[n_t * 32 + n];
  const float bv1 = bias[n_t * 32 + 16 + n];

#pragma unroll
  for (int v = 0; v < 8; ++v) {
    const int m = v + half * 8;
    const float v00 = c00[v] + bv0, v01 = c01[v] + bv1;
    const float v10 = c10[v] + bv0, v11 = c11[v] + bv1;
    const size_t r0 = (size_t)(m_t * 32 + m), r1 = r0 + 16;
    const size_t q0 = (size_t)(n_t * 32 + n), q1 = q0 + 16;
    if constexpr (MODE == 0) {
      uint16_t* o = (uint16_t*)out;
      o[r0 * N + q0] = f2bf(v00);  o[r0 * N + q1] = f2bf(v01);
      o[r1 * N + q0] = f2bf(v10);  o[r1 * N + q1] = f2bf(v11);
    } else if constexpr (MODE == 1) {
      uint16_t* o = (uint16_t*)out;
      o[q0 * M + r0] = f2bf(v00);  o[q1 * M + r0] = f2bf(v01);
      o[q0 * M + r1] = f2bf(v10);  o[q1 * M + r1] = f2bf(v11);
    } else {
      float* o = (float*)out;
      o[r0 * N + q0] = v00;  o[r0 * N + q1] = v01;
      o[r1 * N + q0] = v10;  o[r1 * N + q1] = v11;
    }
  }
}

// ---------- flash attention (one wave per (b, kv, g, 16-query block)) --------
// 64-key blocks per online-softmax update: 8 score WMMAs + 8 PV WMMAs per block,
// one shuffle-reduction chain per 64 keys, masking skipped below the diagonal.
// Q  : bf16 [B*S][768]   Kb : bf16 [B*S][256]   Vt : bf16 [256][B*S]   O : bf16 [B*S][768]
__global__ __launch_bounds__(256, 1)
void attn_wmma(const uint16_t* __restrict__ Q, const uint16_t* __restrict__ Kb,
               const uint16_t* __restrict__ Vt, uint16_t* __restrict__ O) {
  __shared__ uint16_t lds_p[8][16 * 64];   // per-wave P staging (C-layout -> A-frag)

  const int wslot = threadIdx.x >> 5;
  const int wave  = blockIdx.x * 8 + wslot;
  const int qb  = wave & 127;
  int t2 = wave >> 7;
  const int g   = t2 % 3; t2 /= 3;
  const int kvh = t2 & 3;
  const int b   = t2 >> 2;
  const int q0   = qb << 4;
  const int head = kvh * G_ + g;           // repeat_interleave grouping

  const int lid  = threadIdx.x & 31;
  const int half = lid >> 4, nl = lid & 15;

  // loop-invariant Q fragments (d = 0..31, 32..63)
  const uint16_t* qbase = Q + (size_t)(b * S_ + q0) * HID_ + head * D_;
  const v16bf qa0 = load_a_frag(qbase, HID_);
  const v16bf qa1 = load_a_frag(qbase + 32, HID_);

  float rmax[8], rsum[8];
  v8f o0 = {}, o1 = {}, o2 = {}, o3 = {};
#pragma unroll
  for (int v = 0; v < 8; ++v) { rmax[v] = -3.0e38f; rsum[v] = 0.f; }

  uint16_t* myp = &lds_p[wslot][0];
  const int nblk = (q0 + 16 + 63) >> 6;    // 64-key blocks up to the diagonal

  for (int blk = 0; blk < nblk; ++blk) {
    const int t0 = blk << 6;

    // scores: four 16x16 tiles, K-dim = d (64 = 2 wmma steps each)
    const uint16_t* kbb = Kb + (size_t)(b * S_ + t0) * KVD_ + kvh * D_;
    v8f s0 = {}, s1 = {}, s2 = {}, s3 = {};
    s0 = WMMA_BF16(qa0, load_b_frag(kbb + (size_t)0  * KVD_,      KVD_), s0);
    s0 = WMMA_BF16(qa1, load_b_frag(kbb + (size_t)0  * KVD_ + 32, KVD_), s0);
    s1 = WMMA_BF16(qa0, load_b_frag(kbb + (size_t)16 * KVD_,      KVD_), s1);
    s1 = WMMA_BF16(qa1, load_b_frag(kbb + (size_t)16 * KVD_ + 32, KVD_), s1);
    s2 = WMMA_BF16(qa0, load_b_frag(kbb + (size_t)32 * KVD_,      KVD_), s2);
    s2 = WMMA_BF16(qa1, load_b_frag(kbb + (size_t)32 * KVD_ + 32, KVD_), s2);
    s3 = WMMA_BF16(qa0, load_b_frag(kbb + (size_t)48 * KVD_,      KVD_), s3);
    s3 = WMMA_BF16(qa1, load_b_frag(kbb + (size_t)48 * KVD_ + 32, KVD_), s3);

    // scale + (diagonal-only) causal mask + online softmax; stage P into LDS
    const bool needmask = (t0 + 63) > q0;   // wave-uniform
#pragma unroll
    for (int v = 0; v < 8; ++v) {
      const int row = q0 + v + half * 8;
      float x0 = s0[v] * SCALE_;
      float x1 = s1[v] * SCALE_;
      float x2 = s2[v] * SCALE_;
      float x3 = s3[v] * SCALE_;
      if (needmask) {
        if (t0 + nl      > row) x0 = NEGINF_;
        if (t0 + 16 + nl > row) x1 = NEGINF_;
        if (t0 + 32 + nl > row) x2 = NEGINF_;
        if (t0 + 48 + nl > row) x3 = NEGINF_;
      }
      const float bm = rowmax16(fmaxf(fmaxf(x0, x1), fmaxf(x2, x3)));
      const float mn = fmaxf(rmax[v], bm);
      const float alpha = __expf(rmax[v] - mn);
      const float p0 = __expf(x0 - mn);
      const float p1 = __expf(x1 - mn);
      const float p2 = __expf(x2 - mn);
      const float p3 = __expf(x3 - mn);
      rsum[v] = rsum[v] * alpha + rowsum16((p0 + p1) + (p2 + p3));
      rmax[v] = mn;
      o0[v] *= alpha; o1[v] *= alpha; o2[v] *= alpha; o3[v] *= alpha;
      const int row_l = (v + half * 8) * 64;
      myp[row_l + nl]      = f2bf(p0);
      myp[row_l + 16 + nl] = f2bf(p1);
      myp[row_l + 32 + nl] = f2bf(p2);
      myp[row_l + 48 + nl] = f2bf(p3);
    }
    __builtin_amdgcn_wave_barrier();
    asm volatile("s_wait_dscnt 0x0" ::: "memory");   // LDS RAW fence (per-wave)

    const v16bf pa0 = load_a_frag(myp, 64);          // P[:, 0:32]
    const v16bf pa1 = load_a_frag(myp + 32, 64);     // P[:,32:64]

    // PV: B operand rows = d (from V^T), K-dim = keys (64 = 2 wmma steps)
    const uint16_t* vb = Vt + (size_t)(kvh * D_) * MTOT + b * S_ + t0;
    o0 = WMMA_BF16(pa0, load_b_frag(vb + (size_t)0  * MTOT,      MTOT), o0);
    o0 = WMMA_BF16(pa1, load_b_frag(vb + (size_t)0  * MTOT + 32, MTOT), o0);
    o1 = WMMA_BF16(pa0, load_b_frag(vb + (size_t)16 * MTOT,      MTOT), o1);
    o1 = WMMA_BF16(pa1, load_b_frag(vb + (size_t)16 * MTOT + 32, MTOT), o1);
    o2 = WMMA_BF16(pa0, load_b_frag(vb + (size_t)32 * MTOT,      MTOT), o2);
    o2 = WMMA_BF16(pa1, load_b_frag(vb + (size_t)32 * MTOT + 32, MTOT), o2);
    o3 = WMMA_BF16(pa0, load_b_frag(vb + (size_t)48 * MTOT,      MTOT), o3);
    o3 = WMMA_BF16(pa1, load_b_frag(vb + (size_t)48 * MTOT + 32, MTOT), o3);
  }

  // normalize and store O (bf16) for the output projection
  uint16_t* obase = O + (size_t)(b * S_ + q0) * HID_ + head * D_;
#pragma unroll
  for (int v = 0; v < 8; ++v) {
    const float inv = 1.f / rsum[v];
    const int row_l = v + half * 8;
    uint16_t* orow = obase + (size_t)row_l * HID_;
    orow[0  + nl] = f2bf(o0[v] * inv);
    orow[16 + nl] = f2bf(o1[v] * inv);
    orow[32 + nl] = f2bf(o2[v] * inv);
    orow[48 + nl] = f2bf(o3[v] * inv);
  }
}

// ---------- launcher ---------------------------------------------------------

extern "C" void kernel_launch(void* const* d_in, const int* in_sizes, int n_in,
                              void* d_out, int out_size, void* d_ws, size_t ws_size,
                              hipStream_t stream) {
  (void)in_sizes; (void)n_in; (void)out_size; (void)ws_size;
  const float* x  = (const float*)d_in[0];
  const float* Wq = (const float*)d_in[1];
  const float* bq = (const float*)d_in[2];
  const float* Wk = (const float*)d_in[3];
  const float* bk = (const float*)d_in[4];
  const float* Wv = (const float*)d_in[5];
  const float* bv = (const float*)d_in[6];
  const float* Wo = (const float*)d_in[7];
  const float* bo = (const float*)d_in[8];
  float* out = (float*)d_out;

  char* ws = (char*)d_ws;
  uint16_t* Xbf = (uint16_t*)(ws + 0);          // 4096x768
  uint16_t* WqT = (uint16_t*)(ws + 6291456);    // 768x768 (transposed)
  uint16_t* WkT = (uint16_t*)(ws + 7471104);    // 256x768
  uint16_t* WvT = (uint16_t*)(ws + 7864320);    // 256x768
  uint16_t* WoT = (uint16_t*)(ws + 8257536);    // 768x768
  uint16_t* Qbf = (uint16_t*)(ws + 9437184);    // 4096x768
  uint16_t* Kbf = (uint16_t*)(ws + 15728640);   // 4096x256
  uint16_t* Vtb = (uint16_t*)(ws + 17825792);   // 256x4096 (V transposed)
  uint16_t* Obf = (uint16_t*)(ws + 19922944);   // 4096x768

  // 1) precision conversion / weight pre-transpose (all L2-resident)
  cvt_bf16<<<(MTOT * HID_) / 256, 256, 0, stream>>>(x, Xbf, MTOT * HID_);
  transpose_cvt<<<(768 * 768 + 255) / 256, 256, 0, stream>>>(Wq, WqT, 768, 768);
  transpose_cvt<<<(768 * 256 + 255) / 256, 256, 0, stream>>>(Wk, WkT, 768, 256);
  transpose_cvt<<<(768 * 256 + 255) / 256, 256, 0, stream>>>(Wv, WvT, 768, 256);
  transpose_cvt<<<(768 * 768 + 255) / 256, 256, 0, stream>>>(Wo, WoT, 768, 768);

  // 2) QKV projections (bf16 WMMA, f32 accumulate); V stored transposed
  gemm_wmma<0><<<(MTOT / 32) * (768 / 32) / 8, 256, 0, stream>>>(Xbf, WqT, bq, Qbf, MTOT, 768, 768);
  gemm_wmma<0><<<(MTOT / 32) * (256 / 32) / 8, 256, 0, stream>>>(Xbf, WkT, bk, Kbf, MTOT, 256, 768);
  gemm_wmma<1><<<(MTOT / 32) * (256 / 32) / 8, 256, 0, stream>>>(Xbf, WvT, bv, Vtb, MTOT, 256, 768);

  // 3) causal flash attention (2*4*3*128 = 3072 waves)
  attn_wmma<<<(B_ * KV_ * G_ * (S_ / 16)) / 8, 256, 0, stream>>>(Qbf, Kbf, Vtb, Obf);

  // 4) output projection -> f32
  gemm_wmma<2><<<(MTOT / 32) * (768 / 32) / 8, 256, 0, stream>>>(Obf, WoT, bo, out, MTOT, 768, 768);
}
